// FusedEPMoE_63144609186377
// MI455X (gfx1250) — compile-verified, matrix operand order checked by reading, not run on hardware
//
#include <hip/hip_runtime.h>
#include <hip/hip_bf16.h>

typedef __attribute__((ext_vector_type(16))) __bf16 v16bf;
typedef __attribute__((ext_vector_type(8)))  __bf16 v8bf;
typedef __attribute__((ext_vector_type(4)))  __bf16 v4bf;
typedef __attribute__((ext_vector_type(8)))  float  v8f;

#define T_TOK 2048
#define H_DIM 2048
#define F_DIM 1024
#define E_NUM 8

// K-offset inside a 16x32 bf16 A/B fragment (CDNA5 ISA 7.12.2, wave32):
// lane half kg: element pair p<4 -> K = kg*8 + 2p,2p+1 ; p>=4 -> K = 16 + kg*8 + 2(p-4)..
// => per lane the 16 elements are two contiguous runs: [kg*8, kg*8+8) and [16+kg*8, 16+kg*8+8)
__device__ __forceinline__ int frag_k(int p, int kg) {
  return (p < 4) ? (kg * 8 + 2 * p) : (16 + kg * 8 + 2 * (p - 4));
}

// ---------------------------------------------------------------------------
// Kernel 1: top-2 routing. softmax over 8 logits, scatter combine weights,
// build compacted per-expert token lists via atomics.
// ---------------------------------------------------------------------------
__global__ void moe_route_kernel(const float* __restrict__ logits,
                                 float* __restrict__ combine,
                                 int* __restrict__ cnt,
                                 int* __restrict__ list) {
  int t = blockIdx.x * blockDim.x + threadIdx.x;
  if (t >= T_TOK) return;
  float l[E_NUM];
  float m = -3.4e38f;
#pragma unroll
  for (int e = 0; e < E_NUM; ++e) { l[e] = logits[t * E_NUM + e]; m = fmaxf(m, l[e]); }
  float s = 0.f;
#pragma unroll
  for (int e = 0; e < E_NUM; ++e) { l[e] = __expf(l[e] - m); s += l[e]; }
  float inv = 1.0f / s;
  int i1 = 0; float p1 = l[0];
#pragma unroll
  for (int e = 1; e < E_NUM; ++e) if (l[e] > p1) { p1 = l[e]; i1 = e; }
  int i2 = -1; float p2 = -1.f;
#pragma unroll
  for (int e = 0; e < E_NUM; ++e) { if (e == i1) continue; if (l[e] > p2) { p2 = l[e]; i2 = e; } }
#pragma unroll
  for (int e = 0; e < E_NUM; ++e) combine[t * E_NUM + e] = 0.f;
  combine[t * E_NUM + i1] = p1 * inv;
  combine[t * E_NUM + i2] = p2 * inv;
  int s1 = atomicAdd(&cnt[i1], 1); list[i1 * T_TOK + s1] = t;
  int s2 = atomicAdd(&cnt[i2], 1); list[i2 * T_TOK + s2] = t;
}

// ---------------------------------------------------------------------------
// Kernel 2: x (T,H) fp32 -> bf16, row-major (A-side fragments are two
// contiguous 16B runs per lane, so plain row-major bf16 is load-friendly).
// ---------------------------------------------------------------------------
__global__ __launch_bounds__(256)
void x_quant_kernel(const float* __restrict__ x, __bf16* __restrict__ xq) {
  int i = blockIdx.x * blockDim.x + threadIdx.x;   // one float4 per thread
  float4 v = ((const float4*)x)[i];
  v4bf o;
  o[0] = (__bf16)v.x; o[1] = (__bf16)v.y; o[2] = (__bf16)v.z; o[3] = (__bf16)v.w;
  *(v4bf*)(xq + (size_t)i * 4) = o;
}

// ---------------------------------------------------------------------------
// Kernel 3: weight fp32 (E,K,N) -> bf16 pre-swizzled into WMMA B-fragment
// order. One block per 32x16 tile; dst tile = 512 bf16 laid out so lane L of
// a wave32 reads its whole fragment as one contiguous 32B chunk at L*16.
// Tile linear order: ((e*ntiles + nt)*ktiles + kt)  -> inner k loop streams
// sequential 1KB blocks.
// ---------------------------------------------------------------------------
__global__ __launch_bounds__(256)
void w_swizzle_kernel(const float* __restrict__ src, __bf16* __restrict__ dst,
                      int Kdim, int Ndim) {
  __shared__ __bf16 stage[512];
  const int ktiles = Kdim >> 5;
  int bx = blockIdx.x;
  const int kt = bx % ktiles; bx /= ktiles;      // nt, e folded in bx upper bits
  const int tid = threadIdx.x;
  const int n  = tid & 15;                        // column within tile (coalesced reads)
  const int p  = (tid >> 4) & 7;                  // element pair
  const int kg = tid >> 7;                        // lane half
  const int k  = kt * 32 + frag_k(p, kg);
  const float* s = src + (size_t)bx * 32 * 16 * ktiles  // == e*K*N + nt*16 rows-of... see below
                 ;
  // careful: bx now == e*ntiles + nt ; source offset = e*K*N + k*N + nt*16 + n
  {
    const int ntiles = Ndim >> 4;
    const int nt = bx % ntiles;
    const int e  = bx / ntiles;
    s = src + (size_t)e * Kdim * Ndim + (size_t)k * Ndim + nt * 16 + n;
  }
  stage[(kg * 16 + n) * 16 + 2 * p]     = (__bf16)s[0];
  stage[(kg * 16 + n) * 16 + 2 * p + 1] = (__bf16)s[Ndim];
  __syncthreads();
  unsigned* dt = (unsigned*)(dst + (size_t)blockIdx.x * 512);
  dt[tid] = ((const unsigned*)stage)[tid];
}

// ---------------------------------------------------------------------------
// Kernel 4: fused SwiGLU expert FFN over compacted 16-token tiles.
// grid = E * (T/16), block = 256 threads (8 wave32). All operands bf16,
// B fragments one 32B load, A fragments two 16B loads, no conversions.
// ---------------------------------------------------------------------------
__global__ __launch_bounds__(256)
void moe_ffn_kernel(const __bf16* __restrict__ xq,
                    const __bf16* __restrict__ w1q,
                    const __bf16* __restrict__ w3q,
                    const __bf16* __restrict__ w2q,
                    const float* __restrict__ combine,
                    const int* __restrict__ cnt,
                    const int* __restrict__ list,
                    float* __restrict__ out) {
  constexpr int LDA = F_DIM + 8;                 // padded bf16 act row (16B-aligned stride)
  __shared__ __bf16 acts[16 * LDA];              // ~33 KB
  __shared__ int   tok[16];
  __shared__ float cw[16];

  const int e  = blockIdx.x >> 7;                // expert
  const int mt = blockIdx.x & 127;               // 16-token tile index
  const int count = cnt[e];
  if (mt * 16 >= count) return;                  // uniform early exit

  const int tid  = threadIdx.x;
  const int lane = tid & 31;
  const int wv   = tid >> 5;                     // wave id 0..7
  const int kg   = (lane >> 4) & 1;              // lane half
  const int ln16 = lane & 15;

  if (tid < 16) {
    int idx = mt * 16 + tid;
    if (idx < count) {
      int t = list[e * T_TOK + idx];
      tok[tid] = t;
      cw[tid]  = combine[t * E_NUM + e];
    } else { tok[tid] = 0; cw[tid] = 0.f; }      // padded row: weight 0
  }
  __syncthreads();

  const __bf16* __restrict__ arow = xq + (size_t)tok[ln16] * H_DIM + kg * 8;
  const __bf16* __restrict__ w1qe = w1q + (size_t)e * H_DIM * F_DIM;
  const __bf16* __restrict__ w3qe = w3q + (size_t)e * H_DIM * F_DIM;
  const __bf16* __restrict__ w2qe = w2q + (size_t)e * F_DIM * H_DIM;

  // ---- phase 1: act[16,F] = silu(x@w1) * (x@w3) * combine ----
  constexpr int KT1 = H_DIM / 32;                // 64 k-steps
  for (int i = 0; i < 8; ++i) {                  // 64 N-tiles / 8 waves
    const int ntile = wv * 8 + i;
    const __bf16* bgp = w1qe + ((size_t)ntile * KT1) * 512 + lane * 16;
    const __bf16* bup = w3qe + ((size_t)ntile * KT1) * 512 + lane * 16;
    v8f accg = {}; v8f accu = {};
    for (int ks = 0; ks < KT1; ++ks) {
      v16bf a;
      *(v8bf*)&a       = *(const v8bf*)(arow + ks * 32);        // K run [kg*8, +8)
      *((v8bf*)&a + 1) = *(const v8bf*)(arow + ks * 32 + 16);   // K run [16+kg*8, +8)
      v16bf bg = *(const v16bf*)bgp;  bgp += 512;
      v16bf bu = *(const v16bf*)bup;  bup += 512;
      accg = __builtin_amdgcn_wmma_f32_16x16x32_bf16(false, a, false, bg, (short)0, accg, false, false);
      accu = __builtin_amdgcn_wmma_f32_16x16x32_bf16(false, a, false, bu, (short)0, accu, false, false);
    }
#pragma unroll
    for (int r = 0; r < 8; ++r) {                // C layout: M = r + 8*laneHalf
      const int m = (kg << 3) + r;
      float g = accg[r], u = accu[r];
      float v = (g / (1.f + __expf(-g))) * u * cw[m];
      acts[m * LDA + ntile * 16 + ln16] = (__bf16)v;
    }
  }
  __syncthreads();

  // ---- phase 2: y[16,H] += act @ w2, atomic combine into output ----
  constexpr int KT2 = F_DIM / 32;                // 32 k-steps
  const __bf16* arow2 = acts + ln16 * LDA + kg * 8;
  for (int i = 0; i < 16; ++i) {                 // 128 N-tiles / 8 waves
    const int ntile = wv * 16 + i;
    const __bf16* bp = w2qe + ((size_t)ntile * KT2) * 512 + lane * 16;
    v8f acc = {};
    for (int ks = 0; ks < KT2; ++ks) {
      v16bf a;
      *(v8bf*)&a       = *(const v8bf*)(arow2 + ks * 32);       // ds_load_b128
      *((v8bf*)&a + 1) = *(const v8bf*)(arow2 + ks * 32 + 16);
      v16bf b = *(const v16bf*)bp;  bp += 512;
      acc = __builtin_amdgcn_wmma_f32_16x16x32_bf16(false, a, false, b, (short)0, acc, false, false);
    }
#pragma unroll
    for (int r = 0; r < 8; ++r) {
      const int m = (kg << 3) + r;
      atomicAdd(out + (size_t)tok[m] * H_DIM + ntile * 16 + ln16, acc[r]);
    }
  }
}

// ---------------------------------------------------------------------------
extern "C" void kernel_launch(void* const* d_in, const int* in_sizes, int n_in,
                              void* d_out, int out_size, void* d_ws, size_t ws_size,
                              hipStream_t stream) {
  const float* x      = (const float*)d_in[0];   // (T,H)
  const float* logits = (const float*)d_in[1];   // (T,E)
  const float* w1     = (const float*)d_in[2];   // (E,H,F)
  const float* w3     = (const float*)d_in[3];   // (E,H,F)
  const float* w2     = (const float*)d_in[4];   // (E,F,H)
  float* out = (float*)d_out;

  char* ws = (char*)d_ws;
  int*   cnt     = (int*)ws;                               // 8 ints (padded 256B)
  int*   list    = (int*)(ws + 256);                       // E*T ints   = 64KB
  float* combine = (float*)(ws + 256 + E_NUM * T_TOK * 4); // T*E floats = 64KB

  const size_t MB = 1024 * 1024;
  __bf16* xq  = (__bf16*)(ws + 1 * MB);                    //  8 MB
  __bf16* w1q = (__bf16*)(ws + 9 * MB);                    // 32 MB
  __bf16* w3q = (__bf16*)(ws + 41 * MB);                   // 32 MB
  __bf16* w2q = (__bf16*)(ws + 73 * MB);                   // 32 MB (ends at 105 MB)

  hipMemsetAsync(cnt, 0, 256, stream);
  hipMemsetAsync(out, 0, (size_t)out_size * sizeof(float), stream);

  moe_route_kernel<<<T_TOK / 256, 256, 0, stream>>>(logits, combine, cnt, list);

  x_quant_kernel<<<(T_TOK * H_DIM / 4) / 256, 256, 0, stream>>>(x, xq);
  // w1,w3: (E, K=H, N=F) -> E * (F/16) * (H/32) tiles ; w2: (E, K=F, N=H)
  w_swizzle_kernel<<<E_NUM * (F_DIM / 16) * (H_DIM / 32), 256, 0, stream>>>(w1, w1q, H_DIM, F_DIM);
  w_swizzle_kernel<<<E_NUM * (F_DIM / 16) * (H_DIM / 32), 256, 0, stream>>>(w3, w3q, H_DIM, F_DIM);
  w_swizzle_kernel<<<E_NUM * (H_DIM / 16) * (F_DIM / 32), 256, 0, stream>>>(w2, w2q, F_DIM, H_DIM);

  moe_ffn_kernel<<<E_NUM * (T_TOK / 16), 256, 0, stream>>>(xq, w1q, w3q, w2q,
                                                           combine, cnt, list, out);
}
